// TripleModalityFuseMoE_65008624993017
// MI455X (gfx1250) — matrix operand order repaired and checked by reading, not compiled, not run on hardware
//
#include <hip/hip_runtime.h>
#include <math.h>

// ---------------------------------------------------------------------------
// Types for CDNA5 WMMA (wave32)
// ---------------------------------------------------------------------------
typedef __attribute__((ext_vector_type(16))) _Float16 v16h;
typedef __attribute__((ext_vector_type(8)))  float    v8f;

#define WMMA_F16(a, b, c) \
    __builtin_amdgcn_wmma_f32_16x16x32_f16(false, (a), false, (b), (short)0, (c), false, false)

__device__ __forceinline__ float gelu_f(float x) {
    return 0.5f * x * (1.0f + erff(x * 0.70710678118654752f));
}

// Model constants
#define BB    16
#define NW    120
#define CH    27
#define NT    2000
#define HID   256
#define NHEAD 4

// ---------------------------------------------------------------------------
// conv1: implicit GEMM, M=64 (cout), K=189 (27*7, padded 192), stride 4,
// SAME pad (lo=1). Output (1920, 64, 500) f16, exact-erf GELU.
// grid (1920, 8), block 128 (4 waves). Each wave: 16 time positions, 4 M-tiles.
// B tile is staged in LDS pre-im2col'd and pair-packed so each fragment
// register is one aligned 32-bit ds load (no div/mod in the K loop).
// ---------------------------------------------------------------------------
__global__ void conv1_kernel(const float* __restrict__ X,   // (1920, 27, 2000)
                             const float* __restrict__ Wt,  // (64, 27, 7)
                             const float* __restrict__ bias,
                             _Float16* __restrict__ Y) {    // (1920, 64, 500)
    __shared__ _Float16 lw[64 * 192];   // weights, row-major, K padded to 192
    __shared__ _Float16 lb[96 * 128];   // im2col pairs: [kk>>1][t][kk&1], t=0..63
    const int sample = blockIdx.x;
    const int t0g    = blockIdx.y * 64;  // 8 tiles of 64 positions (last masked)
    const int tid    = threadIdx.x;

    for (int i = tid; i < 64 * 192; i += 128) {
        int m = i / 192, kk = i % 192;
        lw[i] = (kk < 189) ? (_Float16)Wt[m * 189 + kk] : (_Float16)0.0f;
    }
    const int pbase = 4 * t0g - 1;       // pad_lo = 1
    const float* xs = X + (size_t)sample * CH * NT;
    for (int i = tid; i < 192 * 64; i += 128) {
        int kk = i >> 6, t = i & 63;
        _Float16 val = (_Float16)0.0f;
        if (kk < 189) {
            int cin = kk / 7, kx = kk - cin * 7;   // constant-divisor, prologue only
            int p = pbase + 4 * t + kx;
            if (p >= 0 && p < NT) val = (_Float16)xs[cin * NT + p];
        }
        lb[(kk >> 1) * 128 + 2 * t + (kk & 1)] = val;
    }
    __syncthreads();

    const int wv = tid >> 5, lane = tid & 31;
    const int ll = lane & 15, lh = lane >> 4;
    const int tloc = wv * 16 + ll;       // 0..63 within tile

    v8f acc[4];
    for (int i = 0; i < 4; ++i)
        for (int j = 0; j < 8; ++j) acc[i][j] = 0.0f;

    for (int kt = 0; kt < 192; kt += 32) {
        v16h bfrag;
#pragma unroll
        for (int v = 0; v < 8; ++v) {
            const _Float16* bp = &lb[((kt >> 1) + v + lh * 8) * 128 + 2 * tloc];
            bfrag[2 * v]     = bp[0];
            bfrag[2 * v + 1] = bp[1];
        }
#pragma unroll
        for (int mt = 0; mt < 4; ++mt) {
            v16h afrag;
            const _Float16* wr = &lw[(mt * 16 + ll) * 192 + kt + lh * 16];
#pragma unroll
            for (int v = 0; v < 8; ++v) {
                int k0 = 2 * v;
                afrag[2 * v]     = wr[k0];
                afrag[2 * v + 1] = wr[k0 + 1];
            }
            acc[mt] = WMMA_F16(afrag, bfrag, acc[mt]);
        }
    }

    const int t = t0g + wv * 16 + ll;
#pragma unroll
    for (int mt = 0; mt < 4; ++mt) {
#pragma unroll
        for (int v = 0; v < 8; ++v) {
            int m = mt * 16 + v + lh * 8;
            float val = gelu_f(acc[mt][v] + bias[m]);
            if (t < 500)
                Y[((size_t)sample * 64 + m) * 500 + t] = (_Float16)val;
        }
    }
}

// ---------------------------------------------------------------------------
// conv2 + GELU + time-mean pool, fused. M=128, K=320 (64*5), N=125 (pad 128),
// stride 4, SAME pad (lo=0). One workgroup per sample; im2col tile built
// directly from the f16 conv1 output into pair-packed LDS.
// Dynamic LDS: (128*320 + 160*256) halves = 160 KB. grid 1920, block 128.
// ---------------------------------------------------------------------------
__global__ void conv2_pool_kernel(const _Float16* __restrict__ H1, // (1920,64,500)
                                  const float* __restrict__ Wt,    // (128,64,5)
                                  const float* __restrict__ bias,
                                  float* __restrict__ P) {         // (1920,128)
    extern __shared__ _Float16 lds[];
    _Float16* lw = lds;                  // 128*320 weights row-major
    _Float16* lb = lds + 128 * 320;      // im2col pairs: [kk>>1][t][kk&1], t=0..127
    __shared__ float accm[128];

    const int sample = blockIdx.x, tid = threadIdx.x;
    for (int i = tid; i < 128 * 320; i += 128) lw[i] = (_Float16)Wt[i];
    const _Float16* hs = H1 + (size_t)sample * 64 * 500;
    for (int i = tid; i < 320 * 128; i += 128) {
        int kk = i >> 7, t = i & 127;
        int cin = kk / 5, kx = kk - cin * 5;       // constant-divisor, prologue only
        int p = 4 * t + kx;                        // pad_lo = 0
        _Float16 val = (p < 500) ? hs[cin * 500 + p] : (_Float16)0.0f;
        lb[(kk >> 1) * 256 + 2 * t + (kk & 1)] = val;
    }
    if (tid < 128) accm[tid] = 0.0f;
    __syncthreads();

    const int wv = tid >> 5, lane = tid & 31;
    const int ll = lane & 15, lh = lane >> 4;

    for (int nt = 0; nt < 8; ++nt) {
        const int tloc = nt * 16 + ll;
        v8f acc[2];
        for (int i = 0; i < 2; ++i)
            for (int j = 0; j < 8; ++j) acc[i][j] = 0.0f;

        for (int kt = 0; kt < 320; kt += 32) {
            v16h bfrag;
#pragma unroll
            for (int v = 0; v < 8; ++v) {
                const _Float16* bp = &lb[((kt >> 1) + v + lh * 8) * 256 + 2 * tloc];
                bfrag[2 * v]     = bp[0];
                bfrag[2 * v + 1] = bp[1];
            }
#pragma unroll
            for (int mi = 0; mi < 2; ++mi) {
                int m = (wv * 2 + mi) * 16 + ll;
                const _Float16* wr = &lw[m * 320 + kt + lh * 16];
                v16h afrag;
#pragma unroll
                for (int v = 0; v < 8; ++v) {
                    int k0 = 2 * v;
                    afrag[2 * v]     = wr[k0];
                    afrag[2 * v + 1] = wr[k0 + 1];
                }
                acc[mi] = WMMA_F16(afrag, bfrag, acc[mi]);
            }
        }
        const int t = nt * 16 + ll;
#pragma unroll
        for (int mi = 0; mi < 2; ++mi) {
#pragma unroll
            for (int v = 0; v < 8; ++v) {
                int m = (wv * 2 + mi) * 16 + v + lh * 8;
                float val = gelu_f(acc[mi][v] + bias[m]);
                if (t < 125) atomicAdd(&accm[m], val);
            }
        }
    }
    __syncthreads();
    if (tid < 128) P[(size_t)sample * 128 + tid] = accm[tid] * (1.0f / 125.0f);
}

// ---------------------------------------------------------------------------
// Generic WMMA GEMM: C[M,N] = act(A[M,K] @ W[K,N] (+bias)).
// fp32 in/out, f16 fragments, f32 accumulate. K % 32 == 0, M,N % 16 == 0.
// grid (ceil(M/64), N/16), block 128 (4 waves; wave w -> M-tile bx*4+w).
// The 32x16 fp32 W slice per k-step is staged coalesced into pair-packed LDS
// and shared by all 4 waves. Inactive tail waves still hit the barriers;
// the 'active' predicate is wave-uniform so EXEC is all-ones at every WMMA.
// flags: bit0 = add bias[n], bit1 = exact GELU.
// ---------------------------------------------------------------------------
__global__ void gemm_wmma_kernel(const float* __restrict__ A,
                                 const float* __restrict__ W,
                                 const float* __restrict__ bias,
                                 float* __restrict__ C,
                                 int M, int K, int N, int flags) {
    __shared__ _Float16 sB[16 * 32];    // pairs: [k>>1][n][k&1], 1 KB
    const int tid = threadIdx.x;
    const int wv = tid >> 5, lane = tid & 31;
    const int ll = lane & 15, lh = lane >> 4;
    const int mt = blockIdx.x * 4 + wv;
    const int nt = blockIdx.y;
    const bool active = (mt * 16 < M);
    const int m = mt * 16 + ll;
    const int n0 = nt * 16;

    v8f acc;
    for (int j = 0; j < 8; ++j) acc[j] = 0.0f;
    const float* ar0 = A + (size_t)(active ? m : 0) * K;

    for (int kt = 0; kt < K; kt += 32) {
        for (int i = tid; i < 512; i += 128) {
            int k = i >> 4, n = i & 15;
            sB[(k >> 1) * 32 + 2 * n + (k & 1)] =
                (_Float16)W[(size_t)(kt + k) * N + n0 + n];
        }
        __syncthreads();
        if (active) {
            v16h a, b;
            const float* ar = ar0 + kt;
#pragma unroll
            for (int v = 0; v < 8; ++v) {
                int k0 = 2 * v + lh * 16;
                a[2 * v]     = (_Float16)ar[k0];
                a[2 * v + 1] = (_Float16)ar[k0 + 1];
            }
#pragma unroll
            for (int v = 0; v < 8; ++v) {
                const _Float16* bp = &sB[(v + lh * 8) * 32 + 2 * ll];
                b[2 * v]     = bp[0];
                b[2 * v + 1] = bp[1];
            }
            acc = WMMA_F16(a, b, acc);
        }
        __syncthreads();
    }
    if (active) {
        const int n = n0 + ll;
#pragma unroll
        for (int v = 0; v < 8; ++v) {
            int row = mt * 16 + v + lh * 8;
            float val = acc[v];
            if (flags & 1) val += bias[n];
            if (flags & 2) val = gelu_f(val);
            C[(size_t)row * N + n] = val;
        }
    }
}

// ---------------------------------------------------------------------------
// Add positional embedding: X[b*120+s, d] += pos[s, d]. grid 1920, block 256.
// ---------------------------------------------------------------------------
__global__ void add_pos_kernel(float* __restrict__ X, const float* __restrict__ pos) {
    int row = blockIdx.x;
    int s = row % NW;
    X[(size_t)row * HID + threadIdx.x] += pos[s * HID + threadIdx.x];
}

// ---------------------------------------------------------------------------
// Aggregator attention (S=120, 4 heads, hd=64, key-padding mask).
// grid 1920 (= b*120+i), block 128 (warp == head).
// ---------------------------------------------------------------------------
__global__ void agg_attn_kernel(const float* __restrict__ Q,
                                const float* __restrict__ K,
                                const float* __restrict__ V,
                                const unsigned char* __restrict__ mask,
                                float* __restrict__ O) {
    __shared__ float sc[NHEAD][NW];
    const int row = blockIdx.x;
    const int b = row / NW;
    const int h = threadIdx.x >> 5;
    const int lane = threadIdx.x & 31;

    const float* qp = Q + (size_t)row * HID + h * 64;
    for (int j = lane; j < NW; j += 32) {
        const float* kp = K + ((size_t)b * NW + j) * HID + h * 64;
        float s = 0.0f;
        for (int d = 0; d < 64; ++d) s += qp[d] * kp[d];
        s *= 0.125f;
        if (mask[b * NW + j]) s = -1e9f;
        sc[h][j] = s;
    }
    __syncthreads();

    float mx = -1e30f;
    for (int j = lane; j < NW; j += 32) mx = fmaxf(mx, sc[h][j]);
    for (int off = 16; off; off >>= 1) mx = fmaxf(mx, __shfl_xor(mx, off, 32));
    float sum = 0.0f;
    for (int j = lane; j < NW; j += 32) {
        float e = expf(sc[h][j] - mx);
        sc[h][j] = e;
        sum += e;
    }
    for (int off = 16; off; off >>= 1) sum += __shfl_xor(sum, off, 32);
    const float inv = 1.0f / sum;
    __syncthreads();

    for (int dd = 0; dd < 2; ++dd) {
        int d = lane + dd * 32;
        float acc = 0.0f;
        for (int j = 0; j < NW; ++j)
            acc += sc[h][j] * V[((size_t)b * NW + j) * HID + h * 64 + d];
        O[(size_t)row * HID + h * 64 + d] = acc * inv;
    }
}

// ---------------------------------------------------------------------------
// Cross-attention over S=3 tokens, 4 heads. grid 16, block 128 (warp == head).
// ---------------------------------------------------------------------------
__global__ void xa_attn_kernel(const float* __restrict__ Q,
                               const float* __restrict__ K,
                               const float* __restrict__ V,
                               float* __restrict__ O) {
    __shared__ float sc[NHEAD][3][3];
    const int b = blockIdx.x;
    const int h = threadIdx.x >> 5;
    const int lane = threadIdx.x & 31;
    if (lane < 9) {
        int i = lane / 3, j = lane % 3;
        const float* qp = Q + ((size_t)b * 3 + i) * HID + h * 64;
        const float* kp = K + ((size_t)b * 3 + j) * HID + h * 64;
        float s = 0.0f;
        for (int d = 0; d < 64; ++d) s += qp[d] * kp[d];
        sc[h][i][j] = s * 0.125f;
    }
    __syncthreads();
    if (lane < 3) {
        float a0 = sc[h][lane][0], a1 = sc[h][lane][1], a2 = sc[h][lane][2];
        float mx = fmaxf(a0, fmaxf(a1, a2));
        float e0 = expf(a0 - mx), e1 = expf(a1 - mx), e2 = expf(a2 - mx);
        float inv = 1.0f / (e0 + e1 + e2);
        sc[h][lane][0] = e0 * inv;
        sc[h][lane][1] = e1 * inv;
        sc[h][lane][2] = e2 * inv;
    }
    __syncthreads();
    for (int dd = 0; dd < 2; ++dd) {
        int d = lane + dd * 32;
        for (int i = 0; i < 3; ++i) {
            float acc = 0.0f;
            for (int j = 0; j < 3; ++j)
                acc += sc[h][i][j] * V[((size_t)b * 3 + j) * HID + h * 64 + d];
            O[((size_t)b * 3 + i) * HID + h * 64 + d] = acc;
        }
    }
}

// ---------------------------------------------------------------------------
// In-place residual + LayerNorm: x = LN(x + r) * g + b. grid rows, block 256.
// ---------------------------------------------------------------------------
__global__ void add_ln_kernel(float* __restrict__ x, const float* __restrict__ r,
                              const float* __restrict__ g, const float* __restrict__ bt) {
    __shared__ float red[HID];
    const int row = blockIdx.x, t = threadIdx.x;
    float v = x[(size_t)row * HID + t] + r[(size_t)row * HID + t];
    red[t] = v;
    __syncthreads();
    for (int s = 128; s; s >>= 1) { if (t < s) red[t] += red[t + s]; __syncthreads(); }
    const float mu = red[0] * (1.0f / HID);
    __syncthreads();
    const float d = v - mu;
    red[t] = d * d;
    __syncthreads();
    for (int s = 128; s; s >>= 1) { if (t < s) red[t] += red[t + s]; __syncthreads(); }
    const float var = red[0] * (1.0f / HID);
    x[(size_t)row * HID + t] = d * rsqrtf(var + 1e-5f) * g[t] + bt[t];
}

// ---------------------------------------------------------------------------
// Masked mean over 120 windows. grid 16, block 256.
// ---------------------------------------------------------------------------
__global__ void mask_mean_kernel(const float* __restrict__ win,
                                 const unsigned char* __restrict__ mask,
                                 float* __restrict__ out) {
    const int b = blockIdx.x, d = threadIdx.x;
    float cnt = 0.0f, s = 0.0f;
    for (int w = 0; w < NW; ++w) {
        float valid = mask[b * NW + w] ? 0.0f : 1.0f;
        cnt += valid;
        s += win[((size_t)b * NW + w) * HID + d] * valid;
    }
    out[b * HID + d] = s / fmaxf(cnt, 1.0f);
}

// ---------------------------------------------------------------------------
// Build (B,3,HID) token stack. grid 16, block 256.
// ---------------------------------------------------------------------------
__global__ void tokens_kernel(const float* __restrict__ tp, const float* __restrict__ ee,
                              const float* __restrict__ sp, const float* __restrict__ tt,
                              const float* __restrict__ et, const float* __restrict__ st,
                              float* __restrict__ T) {
    const int b = blockIdx.x, d = threadIdx.x;
    T[((size_t)b * 3 + 0) * HID + d] = tp[b * HID + d] + tt[d];
    T[((size_t)b * 3 + 1) * HID + d] = ee[b * HID + d] + et[d];
    T[((size_t)b * 3 + 2) * HID + d] = sp[b * HID + d] + st[d];
}

// ---------------------------------------------------------------------------
// MoE gate: softmax over 4 experts, top-2 renormalized. grid 48, block 32.
// ---------------------------------------------------------------------------
__global__ void gate_kernel(const float* __restrict__ x, const float* __restrict__ gw,
                            const float* __restrict__ gb, float* __restrict__ probs,
                            float* __restrict__ topp, int* __restrict__ topi) {
    const int r = blockIdx.x, lane = threadIdx.x;
    float lg[4];
    for (int e = 0; e < 4; ++e) {
        float s = 0.0f;
        for (int d = lane; d < HID; d += 32) s += x[(size_t)r * HID + d] * gw[d * 4 + e];
        for (int off = 16; off; off >>= 1) s += __shfl_xor(s, off, 32);
        lg[e] = s + gb[e];
    }
    if (lane == 0) {
        float mx = fmaxf(fmaxf(lg[0], lg[1]), fmaxf(lg[2], lg[3]));
        float ex[4], sum = 0.0f;
        for (int e = 0; e < 4; ++e) { ex[e] = expf(lg[e] - mx); sum += ex[e]; }
        float p[4];
        for (int e = 0; e < 4; ++e) { p[e] = ex[e] / sum; probs[r * 4 + e] = p[e]; }
        int i0 = 0;
        for (int e = 1; e < 4; ++e) if (p[e] > p[i0]) i0 = e;
        int i1 = -1;
        for (int e = 0; e < 4; ++e) {
            if (e == i0) continue;
            if (i1 < 0 || p[e] > p[i1]) i1 = e;
        }
        float s2 = p[i0] + p[i1];
        topp[r * 2 + 0] = p[i0] / s2;
        topp[r * 2 + 1] = p[i1] / s2;
        topi[r * 2 + 0] = i0;
        topi[r * 2 + 1] = i1;
    }
}

// MoE combine: out = sum_k topp * Eout[topi][r]. grid 48, block 256.
__global__ void moe_combine_kernel(const float* __restrict__ eout,
                                   const float* __restrict__ topp,
                                   const int* __restrict__ topi,
                                   float* __restrict__ out) {
    const int r = blockIdx.x, d = threadIdx.x;
    float acc = 0.0f;
    for (int k = 0; k < 2; ++k) {
        int e = topi[r * 2 + k];
        acc += topp[r * 2 + k] * eout[((size_t)e * 48 + r) * HID + d];
    }
    out[(size_t)r * HID + d] = acc;
}

// Aux loss: usage = probs.mean(0); aux = sum(u*log(1/4) - log(u)/4). block 32.
__global__ void aux_kernel(const float* __restrict__ probs, float* __restrict__ slot) {
    __shared__ float s[4];
    const int e = threadIdx.x;
    if (e < 4) {
        float u = 0.0f;
        for (int r = 0; r < 48; ++r) u += probs[r * 4 + e];
        u *= (1.0f / 48.0f);
        s[e] = u * logf(0.25f) - logf(u) * 0.25f;
    }
    __syncthreads();
    if (e == 0) slot[0] = s[0] + s[1] + s[2] + s[3];
}

// Final: mean over 3 tokens, LN, classifier (256->2), plus aux scalar.
__global__ void final_kernel(const float* __restrict__ T, const float* __restrict__ g,
                             const float* __restrict__ bln, const float* __restrict__ cw,
                             const float* __restrict__ cb, const float* __restrict__ aux,
                             float* __restrict__ out) {
    __shared__ float red[HID];
    __shared__ float xv[HID];
    const int b = blockIdx.x, t = threadIdx.x;
    float v = (T[((size_t)b * 3 + 0) * HID + t] + T[((size_t)b * 3 + 1) * HID + t] +
               T[((size_t)b * 3 + 2) * HID + t]) * (1.0f / 3.0f);
    red[t] = v;
    __syncthreads();
    for (int s = 128; s; s >>= 1) { if (t < s) red[t] += red[t + s]; __syncthreads(); }
    const float mu = red[0] * (1.0f / HID);
    __syncthreads();
    const float d = v - mu;
    red[t] = d * d;
    __syncthreads();
    for (int s = 128; s; s >>= 1) { if (t < s) red[t] += red[t + s]; __syncthreads(); }
    const float var = red[0] * (1.0f / HID);
    xv[t] = d * rsqrtf(var + 1e-5f) * g[t] + bln[t];
    __syncthreads();
    if (t < 2) {
        float acc = 0.0f;
        for (int k = 0; k < HID; ++k) acc += xv[k] * cw[k * 2 + t];
        out[b * 2 + t] = acc + cb[t];
    }
    if (b == 0 && t == 2) out[32] = 0.1f * (aux[0] + aux[1]);
}

// ---------------------------------------------------------------------------
// Host side
// ---------------------------------------------------------------------------
static inline void gemm(const float* A, const float* W, const float* bias, float* C,
                        int M, int K, int N, int flags, hipStream_t s) {
    hipLaunchKernelGGL(gemm_wmma_kernel, dim3((M + 63) / 64, N / 16), dim3(128), 0, s,
                       A, W, bias, C, M, K, N, flags);
}

extern "C" void kernel_launch(void* const* d_in, const int* in_sizes, int n_in,
                              void* d_out, int out_size, void* d_ws, size_t ws_size,
                              hipStream_t stream) {
    (void)in_sizes; (void)n_in; (void)out_size; (void)ws_size;
    auto F = [&](int i) { return (const float*)d_in[i]; };

    // Input index map (setup_inputs dict leaf order)
    const float* text_emb    = F(0);
    const float* eeg_windows = F(1);
    const unsigned char* pmask = (const unsigned char*)d_in[2];
    const float* smiles_emb  = F(3);
    const float* text_W = F(4),  *text_b = F(5);
    const float* c1W = F(6), *c1b = F(7);
    const float* c2W = F(8), *c2b = F(9);
    const float* encW = F(10), *encb = F(11);
    const float* pos = F(12);
    const int AGG0 = 13;                  // 2 layers x 16 leaves -> 13..44
    const float* aggoW = F(45), *aggob = F(46);
    const float* smW = F(47), *smb = F(48);
    const float* tokT = F(49), *tokE = F(50), *tokS = F(51);
    const float* xaWq = F(52), *xabq = F(53), *xaWk = F(54), *xabk = F(55);
    const float* xaWv = F(56), *xabv = F(57), *xaWo = F(58), *xabo = F(59);
    const float* alng = F(60), *alnb = F(61);
    const int MOE0 = 62;                  // 2 layers x 8 leaves -> 62..77
    const float* clsg = F(78), *clsb = F(79), *clsW = F(80), *clsbb = F(81);

    // Workspace carve-up
    size_t off = 0;
    auto alloc = [&](size_t bytes) -> char* {
        size_t o = (off + 255) & ~(size_t)255;
        off = o + bytes;
        return (char*)d_ws + o;
    };
    _Float16* h1 = (_Float16*)alloc((size_t)1920 * 64 * 500 * 2); // conv1 out f16
    float* pooled = (float*)alloc((size_t)1920 * 128 * 4);
    float* X  = (float*)alloc((size_t)1920 * HID * 4);
    float* B1 = (float*)alloc((size_t)1920 * HID * 4);
    float* B2 = (float*)alloc((size_t)1920 * HID * 4);
    float* B3 = (float*)alloc((size_t)1920 * HID * 4);
    float* B4 = (float*)alloc((size_t)1920 * HID * 4);
    float* Bff = (float*)alloc((size_t)1920 * 1024 * 4);
    float* eegmean = (float*)alloc(16 * HID * 4);
    float* eegemb  = (float*)alloc(16 * HID * 4);
    float* tproj   = (float*)alloc(16 * HID * 4);
    float* sproj   = (float*)alloc(16 * HID * 4);
    float* T   = (float*)alloc(48 * HID * 4);
    float* Xq  = (float*)alloc(48 * HID * 4);
    float* Xk  = (float*)alloc(48 * HID * 4);
    float* Xv  = (float*)alloc(48 * HID * 4);
    float* Xc  = (float*)alloc(48 * HID * 4);
    float* Xo  = (float*)alloc(48 * HID * 4);
    float* He  = (float*)alloc(48 * 512 * 4);
    float* Eout = (float*)alloc((size_t)4 * 48 * HID * 4);
    float* Mo  = (float*)alloc(48 * HID * 4);
    float* probs = (float*)alloc(48 * 4 * 4);
    float* topp  = (float*)alloc(48 * 2 * 4);
    int*   topi  = (int*)alloc(48 * 2 * 4);
    float* auxb  = (float*)alloc(2 * 4);

    // --- EEG CNN encoder ---
    hipLaunchKernelGGL(conv1_kernel, dim3(1920, 8), dim3(128), 0, stream,
                       eeg_windows, c1W, c1b, h1);
    const size_t smem2 = (size_t)(128 * 320 + 160 * 256) * 2;   // 160 KB
    hipLaunchKernelGGL(conv2_pool_kernel, dim3(1920), dim3(128), smem2, stream,
                       h1, c2W, c2b, pooled);
    gemm(pooled, encW, encb, X, 1920, 128, HID, 1, stream);
    hipLaunchKernelGGL(add_pos_kernel, dim3(1920), dim3(HID), 0, stream, X, pos);

    // --- Window transformer aggregator (2 layers, post-LN) ---
    for (int L = 0; L < 2; ++L) {
        const int base = AGG0 + 16 * L;
        const float* Wq = F(base + 0),  *bq = F(base + 1);
        const float* Wk = F(base + 2),  *bk = F(base + 3);
        const float* Wv = F(base + 4),  *bv = F(base + 5);
        const float* Wo = F(base + 6),  *bo = F(base + 7);
        const float* l1g = F(base + 8), *l1b = F(base + 9);
        const float* fW1 = F(base + 10), *fb1 = F(base + 11);
        const float* fW2 = F(base + 12), *fb2 = F(base + 13);
        const float* l2g = F(base + 14), *l2b = F(base + 15);

        gemm(X, Wq, bq, B1, 1920, HID, HID, 1, stream);
        gemm(X, Wk, bk, B2, 1920, HID, HID, 1, stream);
        gemm(X, Wv, bv, B3, 1920, HID, HID, 1, stream);
        hipLaunchKernelGGL(agg_attn_kernel, dim3(1920), dim3(128), 0, stream,
                           B1, B2, B3, pmask, B4);
        gemm(B4, Wo, bo, B1, 1920, HID, HID, 1, stream);
        hipLaunchKernelGGL(add_ln_kernel, dim3(1920), dim3(HID), 0, stream,
                           X, B1, l1g, l1b);
        gemm(X, fW1, fb1, Bff, 1920, HID, 1024, 3, stream);   // bias + gelu
        gemm(Bff, fW2, fb2, B1, 1920, 1024, HID, 1, stream);
        hipLaunchKernelGGL(add_ln_kernel, dim3(1920), dim3(HID), 0, stream,
                           X, B1, l2g, l2b);
    }

    // --- Pool + projections + token stack ---
    hipLaunchKernelGGL(mask_mean_kernel, dim3(16), dim3(HID), 0, stream,
                       X, pmask, eegmean);
    gemm(eegmean, aggoW, aggob, eegemb, 16, HID, HID, 1, stream);
    gemm(text_emb, text_W, text_b, tproj, 16, 768, HID, 1, stream);
    gemm(smiles_emb, smW, smb, sproj, 16, 768, HID, 1, stream);
    hipLaunchKernelGGL(tokens_kernel, dim3(16), dim3(HID), 0, stream,
                       tproj, eegemb, sproj, tokT, tokE, tokS, T);

    // --- Cross-modal attention over the 3 tokens ---
    gemm(T, xaWq, xabq, Xq, 48, HID, HID, 1, stream);
    gemm(T, xaWk, xabk, Xk, 48, HID, HID, 1, stream);
    gemm(T, xaWv, xabv, Xv, 48, HID, HID, 1, stream);
    hipLaunchKernelGGL(xa_attn_kernel, dim3(16), dim3(128), 0, stream, Xq, Xk, Xv, Xc);
    gemm(Xc, xaWo, xabo, Xo, 48, HID, HID, 1, stream);
    hipLaunchKernelGGL(add_ln_kernel, dim3(48), dim3(HID), 0, stream, T, Xo, alng, alnb);

    // --- MoE x 2 (dense all-expert compute, top-2 gather) ---
    for (int m = 0; m < 2; ++m) {
        const int base = MOE0 + 8 * m;
        const float* gW = F(base + 0), *gB = F(base + 1);
        const float* W1 = F(base + 2), *b1 = F(base + 3);
        const float* W2 = F(base + 4), *b2 = F(base + 5);
        const float* lg = F(base + 6), *lb = F(base + 7);

        hipLaunchKernelGGL(gate_kernel, dim3(48), dim3(32), 0, stream,
                           T, gW, gB, probs, topp, topi);
        for (int e = 0; e < 4; ++e) {
            gemm(T, W1 + (size_t)e * HID * 512, b1 + e * 512, He,
                 48, HID, 512, 3, stream);                        // gelu
            gemm(He, W2 + (size_t)e * 512 * HID, b2 + e * HID,
                 Eout + (size_t)e * 48 * HID, 48, 512, HID, 1, stream);
        }
        hipLaunchKernelGGL(moe_combine_kernel, dim3(48), dim3(HID), 0, stream,
                           Eout, topp, topi, Mo);
        hipLaunchKernelGGL(aux_kernel, dim3(1), dim3(32), 0, stream, probs, auxb + m);
        hipLaunchKernelGGL(add_ln_kernel, dim3(48), dim3(HID), 0, stream, T, Mo, lg, lb);
    }

    // --- Classifier + aux ---
    hipLaunchKernelGGL(final_kernel, dim3(16), dim3(HID), 0, stream,
                       T, clsg, clsb, clsW, clsbb, auxb, (float*)d_out);
}
